// Self_Att_49658411876624
// MI455X (gfx1250) — matrix operand order repaired and checked by reading, not compiled, not run on hardware
//
#include <hip/hip_runtime.h>
#include <hip/hip_bf16.h>

typedef __bf16 bf16;
typedef bf16  v16bf __attribute__((ext_vector_type(16)));
typedef float v8f   __attribute__((ext_vector_type(8)));

#define C_DIM 1024
#define HEADS 16
#define HD    64
#define NSEQ  2048
#define BATCH 2
#define MROWS (BATCH * NSEQ)   // 4096

// float -> bf16, round to nearest even
__device__ __forceinline__ bf16 f2bf(float f) {
    union { float f; unsigned u; } in; in.f = f;
    unsigned u = in.u;
    unsigned r = (u + 0x7FFFu + ((u >> 16) & 1u)) >> 16;
    union { unsigned short s; bf16 b; } out; out.s = (unsigned short)r;
    return out.b;
}

union FragAB { v16bf v; uint4 u[2]; };

__device__ __forceinline__ v8f wmma_bf16(const FragAB& a, const FragAB& b, v8f c) {
    // D = A(16x32 bf16) * B(32x16 bf16) + C(16x16 f32)
    return __builtin_amdgcn_wmma_f32_16x16x32_bf16(false, a.v, false, b.v,
                                                   (short)0, c, false, false);
}

// async-copy one 64(j) x 32(k) bf16 tile (4 KB) from W[kk..] into LDS.
// 256 threads x 16 bytes each. Wbase = &W[j0*C_DIM + kk]; tracked by ASYNCcnt.
__device__ __forceinline__ void async_wtile(const bf16* Wbase, const bf16* ldsbuf, int t) {
    unsigned lds_off = (unsigned)(size_t)ldsbuf + (unsigned)t * 16u;
    unsigned voff = (unsigned)(((t >> 2) * C_DIM + (t & 3) * 8) * (int)sizeof(bf16));
    unsigned long long sbase = (unsigned long long)(size_t)Wbase;
    asm volatile("global_load_async_to_lds_b128 %0, %1, %2"
                 :: "v"(lds_off), "v"(voff), "s"(sbase) : "memory");
}
__device__ __forceinline__ void wait_async() {
    asm volatile("s_wait_asynccnt 0" ::: "memory");
}

// ---------------------------------------------------------------- f32 -> bf16
__global__ void cvt_kernel(const float* __restrict__ w, bf16* __restrict__ o, int n) {
    int i = blockIdx.x * blockDim.x + threadIdx.x;
    if (i < n) o[i] = f2bf(w[i]);
}

// ------------------------------------------- LayerNorm + positional embedding
__global__ void ln_pos_kernel(const float* __restrict__ x, const float* __restrict__ xpos,
                              const float* __restrict__ g, const float* __restrict__ bb,
                              bf16* __restrict__ xn_bf, bf16* __restrict__ qk_bf) {
    int row = blockIdx.x;
    int tid = threadIdx.x;
    const float* xr = x + (size_t)row * C_DIM;
    float v0[4];
    float s = 0.f, s2 = 0.f;
#pragma unroll
    for (int i = 0; i < 4; ++i) {
        float t = xr[i * 256 + tid];
        v0[i] = t; s += t; s2 += t * t;
    }
    __shared__ float r1[256], r2[256];
    r1[tid] = s; r2[tid] = s2;
    __syncthreads();
    for (int st = 128; st > 0; st >>= 1) {
        if (tid < st) { r1[tid] += r1[tid + st]; r2[tid] += r2[tid + st]; }
        __syncthreads();
    }
    float mu  = r1[0] * (1.f / C_DIM);
    float var = r2[0] * (1.f / C_DIM) - mu * mu;
    float rs  = rsqrtf(var + 1e-5f);
#pragma unroll
    for (int i = 0; i < 4; ++i) {
        int col = i * 256 + tid;
        float xn = (v0[i] - mu) * rs * g[col] + bb[col];
        size_t idx = (size_t)row * C_DIM + col;
        xn_bf[idx] = f2bf(xn);
        qk_bf[idx] = f2bf(xn + xpos[idx]);
    }
}

// ------------------------------------- Q/K/V projections: D[n,j] = A[n,:]·W[j,:]
// Each wave: 32 rows x 64 cols. W k-tiles staged to LDS via async copy (dbl-buffer).
// z==0: Q -> [M,C] bf16   z==1: K -> [M,C] bf16   z==2: V -> transposed [B,H,D,N]
__global__ void proj_kernel(const bf16* __restrict__ qk, const bf16* __restrict__ xn,
                            const bf16* __restrict__ Wq, const bf16* __restrict__ Wk,
                            const bf16* __restrict__ Wv,
                            bf16* __restrict__ Qo, bf16* __restrict__ Ko, bf16* __restrict__ Vt) {
    __shared__ bf16 wtile[2][64 * 32];   // [buf][j][k] 4 KB each
    int tid = threadIdx.x;
    int wid = tid >> 5, lane = tid & 31;
    int z = blockIdx.z;
    const bf16* A = (z == 2) ? xn : qk;
    const bf16* W = (z == 0) ? Wq : (z == 1) ? Wk : Wv;
    int r0 = blockIdx.x * 256 + wid * 32;
    int j0 = blockIdx.y * 64;
    int m    = lane & 15;
    int kb   = (lane >> 4) * 8;    // A-fragment K sub-offset (halfs)
    int kblk = (lane >> 4) * 8;    // B-fragment K offset (halfs)

    v8f acc[2][4];
#pragma unroll
    for (int t = 0; t < 2; ++t)
#pragma unroll
        for (int c = 0; c < 4; ++c)
#pragma unroll
            for (int i = 0; i < 8; ++i) acc[t][c][i] = 0.f;

    const bf16* Wj0 = W + (size_t)j0 * C_DIM;
    async_wtile(Wj0, &wtile[0][0], tid);
    wait_async();
    __syncthreads();

    int cur = 0;
    for (int kk = 0; kk < C_DIM; kk += 32) {
        if (kk + 32 < C_DIM)
            async_wtile(Wj0 + kk + 32, &wtile[cur ^ 1][0], tid);

        // A fragments (global) and all 4 B fragments (LDS) loaded up front so
        // the ds_loads overlap and the 8 WMMAs can issue back-to-back.
        FragAB a[2], bfr[4];
#pragma unroll
        for (int t = 0; t < 2; ++t) {
            const bf16* ap = A + (size_t)(r0 + t * 16 + m) * C_DIM + kk + kb;
            a[t].u[0] = *(const uint4*)ap;
            a[t].u[1] = *(const uint4*)(ap + 16);
        }
#pragma unroll
        for (int c = 0; c < 4; ++c) {
            const bf16* bp = &wtile[cur][(c * 16 + m) * 32 + kblk];
            bfr[c].u[0] = *(const uint4*)bp;
            bfr[c].u[1] = *(const uint4*)(bp + 16);
        }
#pragma unroll
        for (int c = 0; c < 4; ++c) {
            acc[0][c] = wmma_bf16(a[0], bfr[c], acc[0][c]);
            acc[1][c] = wmma_bf16(a[1], bfr[c], acc[1][c]);
        }
        wait_async();
        __syncthreads();
        cur ^= 1;
    }

    int mb = (lane >> 4) * 8;  // D-fragment row base
#pragma unroll
    for (int t = 0; t < 2; ++t) {
        int rbase = r0 + t * 16 + mb;
        if (z < 2) {
            bf16* O = (z == 0) ? Qo : Ko;
#pragma unroll
            for (int c = 0; c < 4; ++c) {
                int j = j0 + c * 16 + m;
#pragma unroll
                for (int r = 0; r < 8; ++r)
                    O[(size_t)(rbase + r) * C_DIM + j] = f2bf(acc[t][c][r]);
            }
        } else {
            int b = rbase >> 11, q = rbase & (NSEQ - 1);
#pragma unroll
            for (int c = 0; c < 4; ++c) {
                int j = j0 + c * 16 + m;
                int h = j >> 6, d = j & 63;
                union { bf16 h8[8]; uint4 u; } pk;
#pragma unroll
                for (int r = 0; r < 8; ++r) pk.h8[r] = f2bf(acc[t][c][r]);
                *(uint4*)(Vt + (size_t)((b * HEADS + h) * HD + d) * NSEQ + q) = pk.u;
            }
        }
    }
}

// ---------------------------------------------------- flash attention (wave32)
// one wave per (b, h, 16-query tile); streams 32 keys/iteration.
// P transpose goes through a private per-wave LDS tile; intra-wave LDS ops are
// in-order (DScnt), so only a compiler scheduling fence is needed (no barrier).
__global__ void attn_kernel(const bf16* __restrict__ Q, const bf16* __restrict__ K,
                            const bf16* __restrict__ Vt, bf16* __restrict__ O) {
    __shared__ float plds[8 * 16 * 32];   // per-wave 16x32 P tile
    int wid = threadIdx.x >> 5, lane = threadIdx.x & 31;
    int tile = blockIdx.x * 8 + wid;      // 4096 tiles total
    int qt = tile & 127;
    int h  = (tile >> 7) & 15;
    int b  = tile >> 11;
    int q0 = qt * 16;
    int m    = lane & 15;
    int kb   = (lane >> 4) * 8;
    int kblk = (lane >> 4) * 8;
    float* pw = plds + wid * 512;

    // Q A-fragments: d-chunks [0,32) and [32,64)
    FragAB aq[2];
    const bf16* qp = Q + (size_t)(b * NSEQ + q0 + m) * C_DIM + h * HD;
#pragma unroll
    for (int t = 0; t < 2; ++t) {
        aq[t].u[0] = *(const uint4*)(qp + t * 32 + kb);
        aq[t].u[1] = *(const uint4*)(qp + t * 32 + kb + 16);
    }

    v8f o[4];
#pragma unroll
    for (int c = 0; c < 4; ++c)
#pragma unroll
        for (int i = 0; i < 8; ++i) o[c][i] = 0.f;
    float rmax[8], rsum[8];
#pragma unroll
    for (int r = 0; r < 8; ++r) { rmax[r] = -1e30f; rsum[r] = 0.f; }

    const bf16* Kbase = K + (size_t)(b * NSEQ) * C_DIM + h * HD;
    const bf16* Vbase = Vt + (size_t)((b * HEADS + h) * HD) * NSEQ;

    for (int kt = 0; kt < NSEQ; kt += 32) {
        // scores for keys [kt, kt+16) and [kt+16, kt+32)
        v8f s0, s1;
        {
            FragAB bk[4];
            const bf16* kp0 = Kbase + (size_t)(kt + m) * C_DIM;
            const bf16* kp1 = Kbase + (size_t)(kt + 16 + m) * C_DIM;
            bk[0].u[0] = *(const uint4*)(kp0 + kblk);
            bk[0].u[1] = *(const uint4*)(kp0 + kblk + 16);
            bk[1].u[0] = *(const uint4*)(kp0 + 32 + kblk);
            bk[1].u[1] = *(const uint4*)(kp0 + 32 + kblk + 16);
            bk[2].u[0] = *(const uint4*)(kp1 + kblk);
            bk[2].u[1] = *(const uint4*)(kp1 + kblk + 16);
            bk[3].u[0] = *(const uint4*)(kp1 + 32 + kblk);
            bk[3].u[1] = *(const uint4*)(kp1 + 32 + kblk + 16);
            v8f zz;
#pragma unroll
            for (int i = 0; i < 8; ++i) zz[i] = 0.f;
            s0 = wmma_bf16(aq[0], bk[0], zz);
            s1 = wmma_bf16(aq[0], bk[2], zz);
            s0 = wmma_bf16(aq[1], bk[1], s0);
            s1 = wmma_bf16(aq[1], bk[3], s1);
        }
        if (kt + 32 < NSEQ)
            __builtin_prefetch(Kbase + (size_t)(kt + 32 + m) * C_DIM + kblk, 0, 1);

        // online softmax; rows live on 16-lane half-groups (lane = key column)
        float alpha[8];
#pragma unroll
        for (int r = 0; r < 8; ++r) {
            float a0 = s0[r] * 0.125f;   // D^-0.5
            float a1 = s1[r] * 0.125f;
            float t = fmaxf(a0, a1);
#pragma unroll
            for (int off = 1; off < 16; off <<= 1)
                t = fmaxf(t, __shfl_xor(t, off, 32));
            float nm = fmaxf(rmax[r], t);
            alpha[r] = __expf(rmax[r] - nm);
            rmax[r] = nm;
            float p0 = __expf(a0 - nm);
            float p1 = __expf(a1 - nm);
            s0[r] = p0; s1[r] = p1;
            float ts = p0 + p1;
#pragma unroll
            for (int off = 1; off < 16; off <<= 1)
                ts += __shfl_xor(ts, off, 32);
            rsum[r] = rsum[r] * alpha[r] + ts;
        }
#pragma unroll
        for (int c = 0; c < 4; ++c)
#pragma unroll
            for (int r = 0; r < 8; ++r) o[c][r] *= alpha[r];

        // P: D-fragment layout -> LDS [16 rows][32 keys] (wave-private region)
        int mb = (lane >> 4) * 8;
#pragma unroll
        for (int r = 0; r < 8; ++r) {
            pw[(mb + r) * 32 + m]      = s0[r];
            pw[(mb + r) * 32 + m + 16] = s1[r];
        }
        __builtin_amdgcn_wave_barrier();   // intra-wave LDS ops are in-order
        // re-read as A-fragment (16x32), convert to bf16
        FragAB apf;
#pragma unroll
        for (int j = 0; j < 8; ++j) {
            apf.v[j]     = f2bf(pw[m * 32 + kb + j]);
            apf.v[j + 8] = f2bf(pw[m * 32 + 16 + kb + j]);
        }
        __builtin_amdgcn_wave_barrier();

        // O += P(16x32) * V(32x64): Vt is [d][n] so B-fragments are contiguous
        FragAB bv[4];
#pragma unroll
        for (int c = 0; c < 4; ++c) {
            const bf16* vp = Vbase + (size_t)(c * 16 + m) * NSEQ + kt + kblk;
            bv[c].u[0] = *(const uint4*)vp;
            bv[c].u[1] = *(const uint4*)(vp + 16);
        }
#pragma unroll
        for (int c = 0; c < 4; ++c)
            o[c] = wmma_bf16(apf, bv[c], o[c]);
    }

    int mb = (lane >> 4) * 8;
#pragma unroll
    for (int r = 0; r < 8; ++r) rsum[r] = 1.f / rsum[r];
#pragma unroll
    for (int c = 0; c < 4; ++c) {
        int j = h * HD + c * 16 + m;
#pragma unroll
        for (int r = 0; r < 8; ++r)
            O[(size_t)(b * NSEQ + q0 + mb + r) * C_DIM + j] = f2bf(o[c][r] * rsum[r]);
    }
}

// ---------------------------- output projection + bias + residual (f32 out)
__global__ void out_kernel(const bf16* __restrict__ Ao, const bf16* __restrict__ Wo,
                           const float* __restrict__ bo, const float* __restrict__ x,
                           float* __restrict__ y) {
    __shared__ bf16 wtile[2][64 * 32];
    int tid = threadIdx.x;
    int wid = tid >> 5, lane = tid & 31;
    int r0 = blockIdx.x * 256 + wid * 32;
    int j0 = blockIdx.y * 64;
    int m    = lane & 15;
    int kb   = (lane >> 4) * 8;
    int kblk = (lane >> 4) * 8;

    v8f acc[2][4];
#pragma unroll
    for (int t = 0; t < 2; ++t)
#pragma unroll
        for (int c = 0; c < 4; ++c)
#pragma unroll
            for (int i = 0; i < 8; ++i) acc[t][c][i] = 0.f;

    const bf16* Wj0 = Wo + (size_t)j0 * C_DIM;
    async_wtile(Wj0, &wtile[0][0], tid);
    wait_async();
    __syncthreads();

    int cur = 0;
    for (int kk = 0; kk < C_DIM; kk += 32) {
        if (kk + 32 < C_DIM)
            async_wtile(Wj0 + kk + 32, &wtile[cur ^ 1][0], tid);

        FragAB a[2], bfr[4];
#pragma unroll
        for (int t = 0; t < 2; ++t) {
            const bf16* ap = Ao + (size_t)(r0 + t * 16 + m) * C_DIM + kk + kb;
            a[t].u[0] = *(const uint4*)ap;
            a[t].u[1] = *(const uint4*)(ap + 16);
        }
#pragma unroll
        for (int c = 0; c < 4; ++c) {
            const bf16* bp = &wtile[cur][(c * 16 + m) * 32 + kblk];
            bfr[c].u[0] = *(const uint4*)bp;
            bfr[c].u[1] = *(const uint4*)(bp + 16);
        }
#pragma unroll
        for (int c = 0; c < 4; ++c) {
            acc[0][c] = wmma_bf16(a[0], bfr[c], acc[0][c]);
            acc[1][c] = wmma_bf16(a[1], bfr[c], acc[1][c]);
        }
        wait_async();
        __syncthreads();
        cur ^= 1;
    }

    int mb = (lane >> 4) * 8;
#pragma unroll
    for (int t = 0; t < 2; ++t) {
#pragma unroll
        for (int c = 0; c < 4; ++c) {
            int j = j0 + c * 16 + m;
            float bj = bo[j];
#pragma unroll
            for (int r = 0; r < 8; ++r) {
                size_t idx = (size_t)(r0 + t * 16 + mb + r) * C_DIM + j;
                y[idx] = acc[t][c][r] + bj + x[idx];
            }
        }
    }
}

extern "C" void kernel_launch(void* const* d_in, const int* in_sizes, int n_in,
                              void* d_out, int out_size, void* d_ws, size_t ws_size,
                              hipStream_t stream) {
    (void)in_sizes; (void)n_in; (void)out_size; (void)ws_size;
    const float* x    = (const float*)d_in[0];
    const float* xpos = (const float*)d_in[1];
    const float* Wq   = (const float*)d_in[2];
    const float* Wk   = (const float*)d_in[3];
    const float* Wv   = (const float*)d_in[4];
    const float* Wo   = (const float*)d_in[5];
    const float* bo   = (const float*)d_in[6];
    const float* lng  = (const float*)d_in[7];
    const float* lnb  = (const float*)d_in[8];
    float* y = (float*)d_out;

    char* ws = (char*)d_ws;
    const size_t MB = 1024 * 1024;
    bf16* qk_bf = (bf16*)(ws);              // 8 MB  [M,C]
    bf16* xn_bf = (bf16*)(ws + 8 * MB);     // 8 MB  [M,C]
    bf16* Wq_bf = (bf16*)(ws + 16 * MB);    // 2 MB
    bf16* Wk_bf = (bf16*)(ws + 18 * MB);    // 2 MB
    bf16* Wv_bf = (bf16*)(ws + 20 * MB);    // 2 MB
    bf16* Wo_bf = (bf16*)(ws + 22 * MB);    // 2 MB
    bf16* Qb    = (bf16*)(ws + 24 * MB);    // 8 MB  [M,C]
    bf16* Kb    = (bf16*)(ws + 32 * MB);    // 8 MB  [M,C]
    bf16* Vt    = (bf16*)(ws + 40 * MB);    // 8 MB  [B,H,D,N]
    bf16* Ob    = (bf16*)(ws + 48 * MB);    // 8 MB  [M,C]

    const int nW = C_DIM * C_DIM;
    cvt_kernel<<<nW / 256, 256, 0, stream>>>(Wq, Wq_bf, nW);
    cvt_kernel<<<nW / 256, 256, 0, stream>>>(Wk, Wk_bf, nW);
    cvt_kernel<<<nW / 256, 256, 0, stream>>>(Wv, Wv_bf, nW);
    cvt_kernel<<<nW / 256, 256, 0, stream>>>(Wo, Wo_bf, nW);

    ln_pos_kernel<<<MROWS, 256, 0, stream>>>(x, xpos, lng, lnb, xn_bf, qk_bf);

    proj_kernel<<<dim3(MROWS / 256, C_DIM / 64, 3), 256, 0, stream>>>(
        qk_bf, xn_bf, Wq_bf, Wk_bf, Wv_bf, Qb, Kb, Vt);

    attn_kernel<<<(BATCH * HEADS * (NSEQ / 16)) / 8, 256, 0, stream>>>(Qb, Kb, Vt, Ob);

    out_kernel<<<dim3(MROWS / 256, C_DIM / 64), 256, 0, stream>>>(Ob, Wo_bf, bo, x, y);
}
